// GnnModel_51754355917461
// MI455X (gfx1250) — matrix operand order repaired and checked by reading, not compiled, not run on hardware
//
#include <hip/hip_runtime.h>

typedef __attribute__((ext_vector_type(16))) _Float16 v16h;
typedef __attribute__((ext_vector_type(8)))  _Float16 v8h;
typedef __attribute__((ext_vector_type(8)))  float    v8f;

static constexpr int N_NODES = 65536;
static constexpr int N_EDGES = 524288;

// ---------------------------------------------------------------------------
// Spline basis: degree-2 open B-spline, kernel_size 3, 2 pseudo-dims -> 9 taps
// ---------------------------------------------------------------------------
__global__ __launch_bounds__(256) void k_basis(const float* __restrict__ pos,
                                               float* __restrict__ basis,
                                               int* __restrict__ widx, int E)
{
    int e = blockIdx.x * blockDim.x + threadIdx.x;
    if (e >= E) return;
    float v0 = pos[2 * e + 0];              // * (KS - DEG) == * 1
    float v1 = pos[2 * e + 1];
    float f0 = v0 - floorf(v0);
    float f1 = v1 - floorf(v1);
    int   b0 = (int)floorf(v0);
    int   b1 = (int)floorf(v1);
    float bx[3], by[3];
    bx[0] = 0.5f * (1.f - f0) * (1.f - f0); bx[1] = -f0 * f0 + f0 + 0.5f; bx[2] = 0.5f * f0 * f0;
    by[0] = 0.5f * (1.f - f1) * (1.f - f1); by[1] = -f1 * f1 + f1 + 0.5f; by[2] = 0.5f * f1 * f1;
    int ix[3], iy[3];
#pragma unroll
    for (int s = 0; s < 3; ++s) {
        int t0 = b0 + s; t0 = t0 < 0 ? 0 : (t0 > 2 ? 2 : t0); ix[s] = t0;
        int t1 = b1 + s; t1 = t1 < 0 ? 0 : (t1 > 2 ? 2 : t1); iy[s] = t1;
    }
#pragma unroll
    for (int sx = 0; sx < 3; ++sx)
#pragma unroll
        for (int sy = 0; sy < 3; ++sy) {
            int s = sx * 3 + sy;
            basis[(size_t)e * 9 + s] = bx[sx] * by[sy];
            widx [(size_t)e * 9 + s] = ix[sx] + 3 * iy[sy];
        }
}

// ---------------------------------------------------------------------------
// Degree (in-degree of dst) -- same for every layer
// ---------------------------------------------------------------------------
__global__ __launch_bounds__(256) void k_deg(const int* __restrict__ ei,
                                             float* __restrict__ deg, int E)
{
    int e = blockIdx.x * blockDim.x + threadIdx.x;
    if (e >= E) return;
    atomicAdd(&deg[ei[E + e]], 1.0f);
}

// ---------------------------------------------------------------------------
// f32 -> f16 activation rows, with optional concat (skip connections) and
// zero K-padding up to Kpad (multiple of 32 for the WMMA K-loop)
// ---------------------------------------------------------------------------
__global__ __launch_bounds__(256) void k_cvt(const float* __restrict__ A, int wa,
                                             const float* __restrict__ Bsrc, int wb,
                                             _Float16* __restrict__ dst, int Kpad, int n)
{
    size_t t = (size_t)blockIdx.x * blockDim.x + threadIdx.x;
    if (t >= (size_t)n * Kpad) return;
    int k   = (int)(t % Kpad);
    int row = (int)(t / Kpad);
    float v = 0.f;
    if (k < wa)                    v = A[(size_t)row * wa + k];
    else if (Bsrc && k < wa + wb)  v = Bsrc[(size_t)row * wb + (k - wa)];
    dst[t] = (_Float16)v;
}

// W[9, ci, co] + root[ci, co] (f32) -> Wt[10*co, Kpad] (f16, K-contiguous,
// zero-padded). Columns [0, 9co): spline kernels; [9co, 10co): root linear,
// so the WMMA GEMM produces the root term for free.
__global__ __launch_bounds__(256) void k_prepw(const float* __restrict__ W,
                                               const float* __restrict__ root,
                                               _Float16* __restrict__ Wt,
                                               int ci, int co, int Kpad)
{
    int ncols = 10 * co;
    int t = blockIdx.x * blockDim.x + threadIdx.x;
    if (t >= ncols * Kpad) return;
    int i = t % Kpad;
    int c = t / Kpad;
    float v = 0.f;
    if (i < ci) {
        if (c < 9 * co) {
            int k = c / co, o = c % co;
            v = W[((size_t)k * ci + i) * co + o];
        } else {
            int o = c - 9 * co;
            v = root[(size_t)i * co + o];
        }
    }
    Wt[(size_t)c * Kpad + i] = (_Float16)v;
}

// lin1_W[32768, 128] (f32) -> Wt[128, 32768] (f16)
__global__ __launch_bounds__(256) void k_prep_lin1(const float* __restrict__ W,
                                                   _Float16* __restrict__ Wt)
{
    size_t t = (size_t)blockIdx.x * blockDim.x + threadIdx.x;
    if (t >= (size_t)128 * 32768) return;
    int c = (int)(t >> 15);
    int i = (int)(t & 32767);
    Wt[t] = (_Float16)W[(size_t)i * 128 + c];
}

// ---------------------------------------------------------------------------
// Fragment load helpers (CDNA5 16-bit A / B VGPR layouts)
// ---------------------------------------------------------------------------
__device__ __forceinline__ v16h load_a_frag(const _Float16* p)
{
    // lane already positioned: chunks at +0 (K j..j+7) and +16 (K j+16..j+23)
    v8h lo = *(const v8h*)(p);
    v8h hi = *(const v8h*)(p + 16);
    v16h a;
#pragma unroll
    for (int j = 0; j < 8; ++j) { a[j] = lo[j]; a[8 + j] = hi[j]; }
    return a;
}

__device__ __forceinline__ v16h load_b_frag(const _Float16* p)
{
    // lane already positioned: chunks at +0 (K j..j+7) and +16 (K j+16..j+23)
    v8h lo = *(const v8h*)(p);
    v8h hi = *(const v8h*)(p + 16);
    v16h b;
#pragma unroll
    for (int j = 0; j < 8; ++j) { b[j] = lo[j]; b[8 + j] = hi[j]; }
    return b;
}

// ---------------------------------------------------------------------------
// WMMA GEMM: C[n_rows, ncols] = A[n_rows, Kpad](f16) x Bt[ncols, Kpad](f16)
// One wave per 32x32 C tile: 2 A x 2 B fragments -> 4 WMMA per K-step,
// each fragment reused twice (2:1 load:wmma ratio).
// ---------------------------------------------------------------------------
__global__ __launch_bounds__(256) void k_gemm(const _Float16* __restrict__ A,
                                              const _Float16* __restrict__ Bt,
                                              float* __restrict__ C,
                                              int n_rows, int Kpad, int ncols)
{
    int wave = (int)(((size_t)blockIdx.x * blockDim.x + threadIdx.x) >> 5);
    int lane = threadIdx.x & 31;
    int nt32  = ncols >> 5;
    int total = (n_rows >> 5) * nt32;
    if (wave >= total) return;           // wave-uniform: EXEC stays all-1s below
    int mt = wave / nt32, nt = wave % nt32;
    int m0 = mt << 5, n0 = nt << 5;
    int half = lane >> 4;                // lane group 0..15 / 16..31
    int l15  = lane & 15;

    const _Float16* ar0 = A  + (size_t)(m0 + l15) * Kpad + (half << 3);
    const _Float16* ar1 = ar0 + (size_t)16 * Kpad;
    const _Float16* br0 = Bt + (size_t)(n0 + l15) * Kpad + (half << 3);
    const _Float16* br1 = br0 + (size_t)16 * Kpad;

    v8f c00 = {}, c01 = {}, c10 = {}, c11 = {};
    for (int k0 = 0; k0 < Kpad; k0 += 32) {
        v16h a0 = load_a_frag(ar0 + k0);
        v16h a1 = load_a_frag(ar1 + k0);
        v16h b0 = load_b_frag(br0 + k0);
        v16h b1 = load_b_frag(br1 + k0);
        c00 = __builtin_amdgcn_wmma_f32_16x16x32_f16(false, a0, false, b0, (short)0, c00, false, false);
        c01 = __builtin_amdgcn_wmma_f32_16x16x32_f16(false, a0, false, b1, (short)0, c01, false, false);
        c10 = __builtin_amdgcn_wmma_f32_16x16x32_f16(false, a1, false, b0, (short)0, c10, false, false);
        c11 = __builtin_amdgcn_wmma_f32_16x16x32_f16(false, a1, false, b1, (short)0, c11, false, false);
    }
    // D layout: VGPR i, lane L -> D[M = i + (L>=16 ? 8 : 0)][N = L&15]
    float* crow = C + (size_t)(m0 + (half << 3)) * ncols + l15;
#pragma unroll
    for (int i = 0; i < 8; ++i) {
        crow[(size_t)i * ncols + n0]             = c00[i];
        crow[(size_t)i * ncols + n0 + 16]        = c01[i];
        crow[(size_t)(i + 16) * ncols + n0]      = c10[i];
        crow[(size_t)(i + 16) * ncols + n0 + 16] = c11[i];
    }
}

// ---------------------------------------------------------------------------
// Edge message passing: thread per (edge, out-channel). 9 gathers from xW
// (L2-resident: xW <= 168 MB < 192 MB L2), fp32 atomic scatter to agg[dst].
// xW row stride is 10*co (last co columns = root term, untouched here).
// ---------------------------------------------------------------------------
__global__ __launch_bounds__(256) void k_gather(const int* __restrict__ ei,
                                                const float* __restrict__ basis,
                                                const int* __restrict__ widx,
                                                const float* __restrict__ xW,
                                                float* __restrict__ agg,
                                                int E, int co_log2)
{
    size_t t = (size_t)blockIdx.x * blockDim.x + threadIdx.x;
    size_t total = (size_t)E << co_log2;
    if (t >= total) return;
    int co = 1 << co_log2;
    int o  = (int)(t & (co - 1));
    int e  = (int)(t >> co_log2);
    int src = ei[e], dst = ei[E + e];
    int ncols = 10 << co_log2;
    const float* xr = xW + (size_t)src * ncols + o;
    const float* bs = basis + (size_t)e * 9;
    const int*   wsi = widx + (size_t)e * 9;
    __builtin_prefetch(xr, 0, 0);        // global_prefetch_b8
    float acc = 0.f;
#pragma unroll
    for (int s = 0; s < 9; ++s)
        acc += bs[s] * xr[(size_t)wsi[s] << co_log2];
    atomicAdd(&agg[(size_t)dst * co + o], acc);
}

// mean-normalize + fused root column + bias + relu6 (pure elementwise now)
__global__ __launch_bounds__(256) void k_final(const float* __restrict__ agg,
                                               const float* __restrict__ deg,
                                               const float* __restrict__ xW,
                                               const float* __restrict__ bias,
                                               float* __restrict__ out,
                                               int co, int n)
{
    size_t t = (size_t)blockIdx.x * blockDim.x + threadIdx.x;
    if (t >= (size_t)n * co) return;
    int o   = (int)(t % co);
    int row = (int)(t / co);
    int ncols = 10 * co;
    float r = bias[o] + xW[(size_t)row * ncols + 9 * co + o];   // x @ root (from WMMA)
    float v = agg[t] / fmaxf(deg[row], 1.0f) + r;
    out[t] = fminf(fmaxf(v, 0.f), 6.f);
}

__global__ __launch_bounds__(256) void k_relu_bias(const float* __restrict__ pre,
                                                   const float* __restrict__ b,
                                                   float* __restrict__ h)
{
    int t = blockIdx.x * blockDim.x + threadIdx.x;
    if (t >= 64 * 128) return;
    h[t] = fmaxf(pre[t] + b[t & 127], 0.f);
}

__global__ __launch_bounds__(192) void k_lin2(const float* __restrict__ h,
                                              const float* __restrict__ W,
                                              const float* __restrict__ b,
                                              float* __restrict__ out)
{
    int t = blockIdx.x * blockDim.x + threadIdx.x;
    if (t >= 64 * 3) return;
    int j = t % 3, bi = t / 3;
    float acc = b[j];
#pragma unroll 16
    for (int c = 0; c < 128; ++c)
        acc += h[bi * 128 + c] * W[c * 3 + j];
    out[t] = acc;
}

// ---------------------------------------------------------------------------
extern "C" void kernel_launch(void* const* d_in, const int* in_sizes, int n_in,
                              void* d_out, int out_size, void* d_ws, size_t ws_size,
                              hipStream_t stream)
{
    (void)in_sizes; (void)n_in; (void)out_size; (void)ws_size;
    const float* x     = (const float*)d_in[0];
    const int*   ei    = (const int*)  d_in[1];
    const float* pos   = (const float*)d_in[2];
    // d_in[3] (batch) is unused by the reference forward pass
    const float* Wc[5] = {(const float*)d_in[4],  (const float*)d_in[7],
                          (const float*)d_in[10], (const float*)d_in[13],
                          (const float*)d_in[16]};
    const float* Rc[5] = {(const float*)d_in[5],  (const float*)d_in[8],
                          (const float*)d_in[11], (const float*)d_in[14],
                          (const float*)d_in[17]};
    const float* Bc[5] = {(const float*)d_in[6],  (const float*)d_in[9],
                          (const float*)d_in[12], (const float*)d_in[15],
                          (const float*)d_in[18]};
    const float* lin1W = (const float*)d_in[19];
    const float* lin1b = (const float*)d_in[20];
    const float* lin2W = (const float*)d_in[21];
    const float* lin2b = (const float*)d_in[22];

    // ---- carve workspace -------------------------------------------------
    char*  ws  = (char*)d_ws;
    size_t off = 0;
    auto carve = [&](size_t bytes) -> void* {
        void* p = ws + off;
        off = (off + bytes + 255) & ~(size_t)255;
        return p;
    };
    float*     basis   = (float*)    carve((size_t)N_EDGES * 9 * 4);
    int*       widx    = (int*)      carve((size_t)N_EDGES * 9 * 4);
    float*     deg     = (float*)    carve((size_t)N_NODES * 4);
    _Float16*  h16     = (_Float16*) carve((size_t)N_NODES * 128 * 2);
    _Float16*  Wt      = (_Float16*) carve((size_t)640 * 128 * 2);
    float*     xW      = (float*)    carve((size_t)N_NODES * 640 * 4);
    float*     agg     = (float*)    carve((size_t)N_NODES * 64 * 4);
    float*     o1      = (float*)    carve((size_t)N_NODES * 32 * 4);
    float*     o2      = (float*)    carve((size_t)N_NODES * 64 * 4);
    float*     o3      = (float*)    carve((size_t)N_NODES * 64 * 4);
    float*     dd0     = (float*)    carve((size_t)N_NODES * 64 * 4);
    float*     dd1     = (float*)    carve((size_t)N_NODES * 32 * 4);
    _Float16*  lin1Wt  = (_Float16*) carve((size_t)128 * 32768 * 2);
    float*     lin1pre = (float*)    carve((size_t)64 * 128 * 4);
    float*     h1      = (float*)    carve((size_t)64 * 128 * 4);

    // ---- shared precompute ----------------------------------------------
    k_basis<<<(N_EDGES + 255) / 256, 256, 0, stream>>>(pos, basis, widx, N_EDGES);
    hipMemsetAsync(deg, 0, (size_t)N_NODES * 4, stream);
    k_deg<<<(N_EDGES + 255) / 256, 256, 0, stream>>>(ei, deg, N_EDGES);

    // ---- 5 SplineConv layers --------------------------------------------
    const int ciA[5] = {16, 32, 64, 128, 96};
    const int coA[5] = {32, 64, 64, 64, 32};
    const int KpA[5] = {32, 32, 64, 128, 96};
    const float* inA[5] = {x,  o1, o2, o3,  dd0};
    const int    waA[5] = {16, 32, 64, 64,  64};
    const float* inB[5] = {nullptr, nullptr, nullptr, o2, o1};
    const int    wbA[5] = {0, 0, 0, 64, 32};
    float*       outA[5] = {o1, o2, o3, dd0, dd1};

    for (int L = 0; L < 5; ++L) {
        int Kpad = KpA[L], co = coA[L], ci = ciA[L];
        int ncols = 10 * co;                      // 9 kernels + fused root
        int clog  = (co == 32) ? 5 : 6;

        size_t ct = (size_t)N_NODES * Kpad;
        k_cvt<<<(unsigned)((ct + 255) / 256), 256, 0, stream>>>(
            inA[L], waA[L], inB[L], wbA[L], h16, Kpad, N_NODES);

        int wtot = ncols * Kpad;
        k_prepw<<<(wtot + 255) / 256, 256, 0, stream>>>(Wc[L], Rc[L], Wt, ci, co, Kpad);

        int tiles = (N_NODES / 32) * (ncols / 32);
        k_gemm<<<(tiles + 7) / 8, 256, 0, stream>>>(h16, Wt, xW, N_NODES, Kpad, ncols);

        hipMemsetAsync(agg, 0, (size_t)N_NODES * co * 4, stream);
        size_t gt = (size_t)N_EDGES << clog;
        k_gather<<<(unsigned)((gt + 255) / 256), 256, 0, stream>>>(
            ei, basis, widx, xW, agg, N_EDGES, clog);

        size_t ft = (size_t)N_NODES * co;
        k_final<<<(unsigned)((ft + 255) / 256), 256, 0, stream>>>(
            agg, deg, xW, Bc[L], outA[L], co, N_NODES);
    }

    // ---- MLP head: flat [64, 32768] (== dd1 in f16) @ lin1_W, relu, lin2 -
    size_t ct = (size_t)N_NODES * 32;
    k_cvt<<<(unsigned)((ct + 255) / 256), 256, 0, stream>>>(
        dd1, 32, nullptr, 0, h16, 32, N_NODES);

    size_t pt = (size_t)128 * 32768;
    k_prep_lin1<<<(unsigned)((pt + 255) / 256), 256, 0, stream>>>(lin1W, lin1Wt);

    int tiles2 = (64 / 32) * (128 / 32);   // 8 tiles -> 1 block
    k_gemm<<<(tiles2 + 7) / 8, 256, 0, stream>>>(h16, lin1Wt, lin1pre, 64, 32768, 128);

    k_relu_bias<<<(64 * 128 + 255) / 256, 256, 0, stream>>>(lin1pre, lin1b, h1);
    k_lin2<<<1, 192, 0, stream>>>(h1, lin2W, lin2b, (float*)d_out);
}
